// SRBMGL2_85366769975907
// MI455X (gfx1250) — compile-verified, matrix operand order checked by reading, not run on hardware
//
#include <hip/hip_runtime.h>

#define B_    1024
#define FV_   4096
#define E_    256
#define Z_    257
#define ZP_   320      // Z padded: multiple of 32 (WMMA K) and 64 (N tile)
#define NS_   16
#define NSTEP_ 8
#define EPS_  1e-3f
#define BSTR_ 40       // LDS B-tile row stride in bf16 (80B: 16B-aligned, bank-spread)

typedef __bf16 bf16x8  __attribute__((ext_vector_type(8)));
typedef __bf16 bf16x16 __attribute__((ext_vector_type(16)));
typedef float  f32x8   __attribute__((ext_vector_type(8)));
typedef int    i32x4   __attribute__((vector_size(16)));   // matches builtin param

#if __has_builtin(__builtin_amdgcn_global_load_async_to_lds_b128)
#define ASYNC_LDS 1
#else
#define ASYNC_LDS 0
#endif

// ---------------- device helpers ----------------

__device__ __forceinline__ float urand_(unsigned a, unsigned b, unsigned c) {
  unsigned x = a * 0x9E3779B1u ^ (b + 0x85EBCA77u) * 0xC2B2AE3Du ^ (c * 0x27D4EB2Fu);
  x ^= x >> 16; x *= 0x7FEB352Du; x ^= x >> 15; x *= 0x846CA68Bu; x ^= x >> 16;
  return (float)(x >> 8) * (1.0f / 16777216.0f);
}
__device__ __forceinline__ float sigmoidf_(float x) { return 1.0f / (1.0f + __expf(-x)); }
__device__ __forceinline__ float logsig_(float x) {
  return x < 0.f ? x - log1pf(__expf(x)) : -log1pf(__expf(-x));
}
__device__ __forceinline__ float softplus_(float x) {
  return x > 0.f ? x + log1pf(__expf(-x)) : log1pf(__expf(x));
}

#if ASYNC_LDS
__device__ __forceinline__ void async_fill_b128(const __bf16* g, __bf16* l) {
  __builtin_amdgcn_global_load_async_to_lds_b128(
      (__attribute__((address_space(1))) i32x4*)(g),
      (__attribute__((address_space(3))) i32x4*)(l), 0, 0);
}
__device__ __forceinline__ void wait_async1() {
#if __has_builtin(__builtin_amdgcn_s_wait_asynccnt)
  __builtin_amdgcn_s_wait_asynccnt(1);
#else
  asm volatile("s_wait_asynccnt 1" ::: "memory");
#endif
}
__device__ __forceinline__ void wait_async0() {
#if __has_builtin(__builtin_amdgcn_s_wait_asynccnt)
  __builtin_amdgcn_s_wait_asynccnt(0);
#else
  asm volatile("s_wait_asynccnt 0" ::: "memory");
#endif
}
#endif

// Load one 16x32 bf16 WMMA fragment from global (row-major, K contiguous).
// ISA 16-bit A/B layout: lanes 0-15 hold K {0..7,16..23}, lanes 16-31 hold
// K {8..15,24..31} of the same 16 rows.
__device__ __forceinline__ bf16x16 load_frag_g(const __bf16* base, int ld, int row0,
                                               int k0, int lane) {
  const int half = lane >> 4, r = lane & 15;
  const __bf16* p = base + (size_t)(row0 + r) * ld + k0 + half * 8;
  bf16x8 lo = *(const bf16x8*)(p);
  bf16x8 hi = *(const bf16x8*)(p + 16);
  bf16x16 f;
#pragma unroll
  for (int i = 0; i < 8; i++) { f[i] = lo[i]; f[i + 8] = hi[i]; }
  return f;
}

// Same fragment shape out of the LDS-staged 64xBSTR_ tile (ds_load_b128 x2).
__device__ __forceinline__ bf16x16 load_frag_l(const __bf16* base, int row0, int lane) {
  const int half = lane >> 4, r = lane & 15;
  const __bf16* p = base + (row0 + r) * BSTR_ + half * 8;
  bf16x8 lo = *(const bf16x8*)(p);
  bf16x8 hi = *(const bf16x8*)(p + 16);
  bf16x16 f;
#pragma unroll
  for (int i = 0; i < 8; i++) { f[i] = lo[i]; f[i + 8] = hi[i]; }
  return f;
}

// ---------------- fused WMMA GEMM ----------------
// C[m,n] = sum_k A[m,k] * Bt[n,k]  (+ bias[n]), then mode-specific epilogue.
// MODE 0: r   = (img - (C+bias)) * exp(2*b_scale)        -> bf16 outB [M,N]
// MODE 1: dh1 = C                                        -> f32 outF
// MODE 2: hpl = C+bias -> outF;  t = node_h*sigmoid(-hpl)-> bf16 stateB
// MODE 3: dz1 = C+bias; z_next=sigmoid; stochastic node_z update (stateF/B)
// MODE 4: h_noise = Bernoulli(sigmoid(C+bias))           -> bf16 outB
// MODE 5: Normal log-prob of img vs (C+bias), row-reduced, atomicAdd joint[m]
template <int MODE>
__global__ __launch_bounds__(256) void gemm_wmma_bf16(
    const __bf16* __restrict__ A, int lda, const __bf16* __restrict__ Bt, int ldb,
    int M, int N, int K, const float* __restrict__ bias,
    const float* __restrict__ img, const float* __restrict__ bsp,
    const float* __restrict__ auxF, float* __restrict__ outF,
    __bf16* __restrict__ outB, float* __restrict__ stateF,
    __bf16* __restrict__ stateB, float* __restrict__ joint, int nlim,
    unsigned seed) {
  __shared__ __bf16 Bs[2][64][BSTR_];
  const int tid = threadIdx.x;
  const int lane = tid & 31;
  const int wave = tid >> 5;
  const int n0 = blockIdx.x * 64;
  const int m0 = blockIdx.y * 128 + wave * 16;
  const int nl = lane & 15;
  const int rb = (lane >> 4) << 3;
  // cooperative B-stage: each thread owns one 16B chunk of the 64x32 tile
  const int srow = tid >> 2;        // 0..63
  const int schk = (tid & 3) * 8;   // bf16 element offset: 0,8,16,24

  f32x8 acc[4];
#pragma unroll
  for (int t = 0; t < 4; t++)
#pragma unroll
    for (int j = 0; j < 8; j++) acc[t][j] = 0.0f;

  const int nsteps = K / 32;
#if ASYNC_LDS
  // ---- async double-buffered pipeline: fill(k+1) overlaps WMMA on k ----
  async_fill_b128(Bt + (size_t)(n0 + srow) * ldb + schk, &Bs[0][srow][schk]);
  for (int s = 0; s < nsteps; ++s) {
    const int k0 = s * 32;
    const int cur = s & 1;
    if (s + 1 < nsteps) {
      async_fill_b128(Bt + (size_t)(n0 + srow) * ldb + (k0 + 32) + schk,
                      &Bs[cur ^ 1][srow][schk]);
      wait_async1();   // in-order: slice k is resident, k+1 still in flight
    } else {
      wait_async0();
    }
    __syncthreads();
    bf16x16 af = load_frag_g(A, lda, m0, k0, lane);
    bf16x16 bfr[4];
#pragma unroll
    for (int t = 0; t < 4; t++)   // issue all 8 ds_load_b128 first ...
      bfr[t] = load_frag_l(&Bs[cur][0][0], t * 16, lane);
#pragma unroll
    for (int t = 0; t < 4; t++)   // ... then a clean back-to-back WMMA burst
      acc[t] = __builtin_amdgcn_wmma_f32_16x16x32_bf16(
          false, af, false, bfr[t], (short)0, acc[t], false, false);
    __syncthreads();   // reads done before this buffer is refilled
  }
#else
  // ---- fallback: synchronous cooperative staging (global->reg->LDS) ----
  for (int s = 0; s < nsteps; ++s) {
    const int k0 = s * 32;
    bf16x8 stage = *(const bf16x8*)(Bt + (size_t)(n0 + srow) * ldb + k0 + schk);
    __syncthreads();
    *(bf16x8*)&Bs[0][srow][schk] = stage;
    __syncthreads();
    bf16x16 af = load_frag_g(A, lda, m0, k0, lane);
    bf16x16 bfr[4];
#pragma unroll
    for (int t = 0; t < 4; t++)
      bfr[t] = load_frag_l(&Bs[0][0][0], t * 16, lane);
#pragma unroll
    for (int t = 0; t < 4; t++)
      acc[t] = __builtin_amdgcn_wmma_f32_16x16x32_bf16(
          false, af, false, bfr[t], (short)0, acc[t], false, false);
  }
#endif

  if (MODE == 5) {
    const float bs = bsp[0];
    const float inv2 = __expf(-2.0f * bs);
#pragma unroll
    for (int r = 0; r < 8; r++) {
      const int m = m0 + rb + r;
      const size_t mb = (size_t)(m & nlim) * N;  // nlim = B-1 mask -> batch row
      float s = 0.0f;
#pragma unroll
      for (int t = 0; t < 4; t++) {
        const int n = n0 + t * 16 + nl;
        const float v = acc[t][r] + bias[n];
        const float d = img[mb + n] - v;
        s += -0.5f * d * d * inv2;
      }
      // reduce the 16 N-columns of this row (one 16-lane group)
#pragma unroll
      for (int off = 8; off; off >>= 1) s += __shfl_xor(s, off, 32);
      if (nl == 0) atomicAdd(&joint[m], s);
    }
    return;
  }

  float out_prec = 0.0f;
  if (MODE == 0) out_prec = __expf(2.0f * bsp[0]);
#pragma unroll
  for (int t = 0; t < 4; t++) {
    const int n = n0 + t * 16 + nl;
    const float bv = bias ? bias[n] : 0.0f;
#pragma unroll
    for (int r = 0; r < 8; r++) {
      const int m = m0 + rb + r;
      const size_t idx = (size_t)m * N + n;
      const float v = acc[t][r] + bv;
      if (MODE == 0) {
        outB[idx] = (__bf16)((img[idx] - v) * out_prec);
      } else if (MODE == 1) {
        outF[idx] = v;
      } else if (MODE == 2) {
        outF[idx] = v;
        stateB[idx] = (__bf16)(auxF[idx] * sigmoidf_(-v));
      } else if (MODE == 3) {
        if (n < nlim) {
          const float u = urand_(seed, (unsigned)m, (unsigned)n);
          if (u < 0.5f) {
            const float zn = sigmoidf_(v);
            const float nz = 0.5f * (stateF[idx] + zn);
            stateF[idx] = nz;
            stateB[idx] = (__bf16)nz;
          }
        }
      } else if (MODE == 4) {
        const float u = urand_(seed, (unsigned)m, (unsigned)n);
        outB[idx] = (__bf16)((u < sigmoidf_(v)) ? 1.0f : 0.0f);
      }
    }
  }
}

// ---------------- pack weights + init state ----------------
__global__ void pack_init_kernel(const float* __restrict__ wdec,
                                 const float* __restrict__ whz,
                                 const float* __restrict__ blp,
                                 __bf16* wdec_bf, __bf16* wdecT_bf, __bf16* whz_bf,
                                 __bf16* whzT_bf, float* lps, float* nh_f,
                                 __bf16* nh_bf, float* nz_f, __bf16* nz_bf) {
  const size_t stride = (size_t)gridDim.x * blockDim.x;
  for (size_t i = (size_t)blockIdx.x * blockDim.x + threadIdx.x;
       i < (size_t)FV_ * E_; i += stride) {
    const float w = wdec[i];
    wdec_bf[i] = (__bf16)w;
    const size_t f = i / E_, e = i % E_;
    wdecT_bf[e * FV_ + f] = (__bf16)w;                    // [E, FV]
    if (i < (size_t)E_ * ZP_) {
      const size_t e2 = i / ZP_, z = i % ZP_;
      whz_bf[i] = (__bf16)(z < Z_ ? whz[e2 * Z_ + z] : 0.0f);   // [E, ZP]
      const size_t n = i / E_, e3 = i % E_;
      whzT_bf[i] = (__bf16)(n < Z_ ? whz[e3 * Z_ + n] : 0.0f);  // [ZP, E]
    }
    if (i < ZP_) lps[i] = (i < Z_) ? logsig_(blp[i]) : 0.0f;
    if (i < (size_t)B_ * E_) { nh_f[i] = EPS_; nh_bf[i] = (__bf16)EPS_; }
    if (i < (size_t)B_ * ZP_) {
      const size_t z = i % ZP_;
      const float v = (z < Z_) ? EPS_ : 0.0f;   // pad must be 0 for K-dim GEMMs
      nz_f[i] = v; nz_bf[i] = (__bf16)v;
    }
  }
}

// ---------------- node_h stochastic update ----------------
__global__ void update_h_kernel(const float* __restrict__ dh1,
                                const float* __restrict__ hpl, float* nh_f,
                                __bf16* nh_bf, unsigned seed) {
  const size_t stride = (size_t)gridDim.x * blockDim.x;
  for (size_t i = (size_t)blockIdx.x * blockDim.x + threadIdx.x;
       i < (size_t)B_ * E_; i += stride) {
    const float hn = sigmoidf_(dh1[i] + logsig_(hpl[i]));
    if (urand_(seed, (unsigned)i, 0x68u) < 0.5f) {
      const float h = 0.5f * (nh_f[i] + hn);
      nh_f[i] = h; nh_bf[i] = (__bf16)h;
    }
  }
}

// ---------------- z sampling + lp_internal + lp_ext constant ----------------
__global__ void sample_z_kernel(const float* __restrict__ nz_f,
                                const float* __restrict__ blp,
                                const float* __restrict__ bsp,
                                __bf16* __restrict__ z_noise, float* joint) {
  const int i = blockIdx.x * blockDim.x + threadIdx.x;  // i = s*B + b
  if (i >= NS_ * B_) return;
  const int b = i & (B_ - 1);
  const float bs = bsp[0];
  float acc = (float)FV_ * (-bs - 0.9189385332046727f);  // sum of lp_ext consts
  for (int z = 0; z < ZP_; z++) {
    __bf16 out = (__bf16)0.0f;
    if (z < Z_) {
      const float pz = nz_f[(size_t)b * ZP_ + z];
      const float zn = (urand_(0x5151u, (unsigned)i, (unsigned)z) < pz) ? 1.0f : 0.0f;
      out = (__bf16)zn;
      const float bl = blp[z];
      acc += zn * bl - softplus_(bl);                         // lp_prior
      acc -= zn * __logf(pz) + (1.0f - zn) * log1pf(-pz);     // -lp_post
    }
    z_noise[(size_t)i * ZP_ + z] = out;
  }
  joint[i] = acc;
}

// ---------------- logsumexp over NS, mean over B ----------------
__global__ void final_kernel(const float* __restrict__ joint, float* out) {
  __shared__ float red[256];
  float local = 0.0f;
  for (int b = threadIdx.x; b < B_; b += 256) {
    float mx = -3.0e38f;
    for (int s = 0; s < NS_; s++) mx = fmaxf(mx, joint[s * B_ + b]);
    float se = 0.0f;
    for (int s = 0; s < NS_; s++) se += __expf(joint[s * B_ + b] - mx);
    local += mx + __logf(se) - 2.772588722239781f;  // - log(NS)
  }
  red[threadIdx.x] = local;
  __syncthreads();
  for (int off = 128; off; off >>= 1) {
    if (threadIdx.x < off) red[threadIdx.x] += red[threadIdx.x + off];
    __syncthreads();
  }
  if (threadIdx.x == 0) out[0] = -red[0] / (float)B_;
}

// ---------------- host ----------------
extern "C" void kernel_launch(void* const* d_in, const int* in_sizes, int n_in,
                              void* d_out, int out_size, void* d_ws, size_t ws_size,
                              hipStream_t stream) {
  (void)in_sizes; (void)n_in; (void)out_size; (void)ws_size;
  const float* img    = (const float*)d_in[0];
  // d_in[1] = lab (int64) — unused by the loss
  const float* wdec   = (const float*)d_in[2];
  const float* wdec_b = (const float*)d_in[3];
  const float* whz    = (const float*)d_in[4];
  const float* whz_b  = (const float*)d_in[5];
  const float* blp    = (const float*)d_in[6];
  const float* bsp    = (const float*)d_in[7];

  char* ws = (char*)d_ws;
  size_t off = 0;
  auto alloc = [&](size_t bytes) -> void* {
    off = (off + 255) & ~(size_t)255;
    void* p = ws + off;
    off += bytes;
    return p;
  };
  __bf16* wdec_bf  = (__bf16*)alloc((size_t)FV_ * E_ * 2);
  __bf16* wdecT_bf = (__bf16*)alloc((size_t)E_ * FV_ * 2);
  __bf16* whz_bf   = (__bf16*)alloc((size_t)E_ * ZP_ * 2);
  __bf16* whzT_bf  = (__bf16*)alloc((size_t)ZP_ * E_ * 2);
  float*  lps      = (float*)alloc((size_t)ZP_ * 4);
  float*  nh_f     = (float*)alloc((size_t)B_ * E_ * 4);
  __bf16* nh_bf    = (__bf16*)alloc((size_t)B_ * E_ * 2);
  float*  nz_f     = (float*)alloc((size_t)B_ * ZP_ * 4);
  __bf16* nz_bf    = (__bf16*)alloc((size_t)B_ * ZP_ * 2);
  __bf16* r_bf     = (__bf16*)alloc((size_t)B_ * FV_ * 2);
  float*  dh1_f    = (float*)alloc((size_t)B_ * E_ * 4);
  float*  hpl_f    = (float*)alloc((size_t)B_ * E_ * 4);
  __bf16* t_bf     = (__bf16*)alloc((size_t)B_ * E_ * 2);
  __bf16* z_noise  = (__bf16*)alloc((size_t)NS_ * B_ * ZP_ * 2);
  __bf16* h_noise  = (__bf16*)alloc((size_t)NS_ * B_ * E_ * 2);
  float*  joint    = (float*)alloc((size_t)NS_ * B_ * 4);

  pack_init_kernel<<<4096, 256, 0, stream>>>(wdec, whz, blp, wdec_bf, wdecT_bf,
                                             whz_bf, whzT_bf, lps, nh_f, nh_bf,
                                             nz_f, nz_bf);

  for (int step = 0; step < NSTEP_; step++) {
    // r = (img - (h @ Wdec^T + b)) * exp(2*bs)        [B,FV] bf16
    gemm_wmma_bf16<0><<<dim3(FV_ / 64, B_ / 128), 256, 0, stream>>>(
        nh_bf, E_, wdec_bf, E_, B_, FV_, E_, wdec_b, img, bsp, nullptr, nullptr,
        r_bf, nullptr, nullptr, nullptr, 0, 0u);
    // dh1 = r @ Wdec                                  [B,E] f32
    gemm_wmma_bf16<1><<<dim3(E_ / 64, B_ / 128), 256, 0, stream>>>(
        r_bf, FV_, wdecT_bf, FV_, B_, E_, FV_, nullptr, nullptr, nullptr, nullptr,
        dh1_f, nullptr, nullptr, nullptr, nullptr, 0, 0u);
    // hpl = z @ Whz^T + b;  t = h * sigmoid(-hpl)     [B,E]
    gemm_wmma_bf16<2><<<dim3(E_ / 64, B_ / 128), 256, 0, stream>>>(
        nz_bf, ZP_, whz_bf, ZP_, B_, E_, ZP_, whz_b, nullptr, nullptr, nh_f,
        hpl_f, nullptr, nullptr, t_bf, nullptr, 0, 0u);
    // dz1 = t @ Whz + log_prior_sig; fused node_z update   [B,ZP]
    gemm_wmma_bf16<3><<<dim3(ZP_ / 64, B_ / 128), 256, 0, stream>>>(
        t_bf, E_, whzT_bf, E_, B_, ZP_, E_, lps, nullptr, nullptr, nullptr,
        nullptr, nullptr, nz_f, nz_bf, nullptr, Z_, 1000u + (unsigned)step);
    update_h_kernel<<<(B_ * E_ + 255) / 256, 256, 0, stream>>>(
        dh1_f, hpl_f, nh_f, nh_bf, 2000u + (unsigned)step);
  }

  // z_noise Bernoulli + lp_internal sum -> joint[s,b]
  sample_z_kernel<<<(NS_ * B_ + 255) / 256, 256, 0, stream>>>(nz_f, blp, bsp,
                                                              z_noise, joint);
  // h_noise = Bernoulli(sigmoid(z_noise @ Whz^T + b))      [NS*B, E]
  gemm_wmma_bf16<4><<<dim3(E_ / 64, (NS_ * B_) / 128), 256, 0, stream>>>(
      z_noise, ZP_, whz_bf, ZP_, NS_ * B_, E_, ZP_, whz_b, nullptr, nullptr,
      nullptr, nullptr, h_noise, nullptr, nullptr, nullptr, 0, 777u);
  // lp_external reduced over FV, accumulated into joint[s,b]
  gemm_wmma_bf16<5><<<dim3(FV_ / 64, (NS_ * B_) / 128), 256, 0, stream>>>(
      h_noise, E_, wdec_bf, E_, NS_ * B_, FV_, E_, wdec_b, img, bsp, nullptr,
      nullptr, nullptr, nullptr, nullptr, joint, B_ - 1, 0u);

  final_kernel<<<1, 256, 0, stream>>>(joint, (float*)d_out);
}
